// LiteMoE_44616120270876
// MI455X (gfx1250) — matrix operand order repaired and checked by reading, not compiled, not run on hardware
//
#include <hip/hip_runtime.h>
#include <hip/hip_bf16.h>

// MoE: E=8 experts, top-2, H=1024, I=2048, T=2048 tokens, fp32 in/out.
// Pipeline:
//   1) cvt_kernel x4  : w1/w3/w2/x fp32 -> bf16 into workspace (memory-bound)
//   2) router_kernel  : dense per-token expert weights ew[T,8] into workspace
//   3) moe_kernel     : fused SwiGLU experts, M=32 register blocking,
//                       v_wmma_f32_16x16x32_bf16, bf16 B-fragments from memory,
//                       x-tile staged to LDS via global_load_async_to_lds_b128.

#define E_ 8
#define H_ 1024
#define I_ 2048
#define T_ 2048

typedef __attribute__((ext_vector_type(16))) __bf16        v16bf;
typedef __attribute__((ext_vector_type(8)))  float         v8f;
typedef __attribute__((ext_vector_type(4)))  float         v4f;
typedef __attribute__((ext_vector_type(4)))  unsigned int  v4u;

// fp32 -> bf16 with round-to-nearest-even
__device__ __forceinline__ unsigned short f2bf(float f) {
  unsigned int u = __float_as_uint(f);
  return (unsigned short)((u + 0x7FFFu + ((u >> 16) & 1u)) >> 16);
}

// silu via hardware transcendentals: x * rcp(1 + exp(-x))
__device__ __forceinline__ float silu_f(float x) {
  return x * __builtin_amdgcn_rcpf(1.0f + __expf(-x));
}

union BF16x16 { unsigned short s[16]; v4u q[2]; v16bf v; };

// A-fragment (16x32 bf16) from row-major LDS tile.
// Lane L (L<16): M=L, K = {0..7, 16..23}; lane L>=16: M=L-16, K = {8..15, 24..31}.
// koff already includes kc + (lane>>4)*8.
__device__ __forceinline__ v16bf load_afrag_lds(const unsigned short* base,
                                                int stride, int row, int koff) {
  const unsigned short* p = base + row * stride + koff;
  BF16x16 u;
  u.q[0] = *(const v4u*)p;          // K .. K+7
  u.q[1] = *(const v4u*)(p + 16);   // K+16 .. K+23
  return u.v;
}

// B-fragment (32x16 bf16) from pre-converted bf16 weights: lane n<16 holds
// column n, K = {kb..kb+7, kb+16..kb+23}; lanes>=16 same columns at K+8 (caller
// offsets pointer). Two global_load_b128, no VALU cvt.
__device__ __forceinline__ v16bf load_bfrag_bf16(const unsigned short* p) {
  BF16x16 u;
  u.q[0] = *(const v4u*)p;
  u.q[1] = *(const v4u*)(p + 16);
  return u.v;
}

__device__ __forceinline__ v8f wmma_bf16(v16bf a, v16bf b, v8f c) {
  // (neg_a, A, neg_b, B, c_mod, C, reuse_a, reuse_b)
  return __builtin_amdgcn_wmma_f32_16x16x32_bf16(false, a, false, b,
                                                 (short)0, c, false, false);
}

// ---------------- fp32 -> bf16 bulk converter (8 elements / thread) ---------
__global__ __launch_bounds__(256) void cvt_kernel(const float* __restrict__ src,
                                                  unsigned short* __restrict__ dst) {
  size_t i = ((size_t)blockIdx.x * 256 + threadIdx.x) * 8;
  v4f a = *(const v4f*)(src + i);
  v4f b = *(const v4f*)(src + i + 4);
  v4u o;
  o[0] = (unsigned)f2bf(a[0]) | ((unsigned)f2bf(a[1]) << 16);
  o[1] = (unsigned)f2bf(a[2]) | ((unsigned)f2bf(a[3]) << 16);
  o[2] = (unsigned)f2bf(b[0]) | ((unsigned)f2bf(b[1]) << 16);
  o[3] = (unsigned)f2bf(b[2]) | ((unsigned)f2bf(b[3]) << 16);
  *(v4u*)(dst + i) = o;
}

// ---------------- Router: logits = x @ gate_w^T, softmax, top-2, renorm -----
__global__ __launch_bounds__(256) void router_kernel(
    const float* __restrict__ x, const float* __restrict__ gate_w,
    float* __restrict__ ew) {
  __shared__ float gw[E_ * H_];  // 32 KB
  for (int i = threadIdx.x; i < E_ * H_; i += 256) gw[i] = gate_w[i];
  __syncthreads();

  int t = blockIdx.x * 256 + threadIdx.x;
  const float* xp = x + (size_t)t * H_;
  float acc[E_];
#pragma unroll
  for (int e = 0; e < E_; e++) acc[e] = 0.0f;

  for (int h = 0; h < H_; h += 4) {
    v4f xv = *(const v4f*)(xp + h);
#pragma unroll
    for (int e = 0; e < E_; e++) {
      acc[e] += xv[0] * gw[e * H_ + h]     + xv[1] * gw[e * H_ + h + 1]
              + xv[2] * gw[e * H_ + h + 2] + xv[3] * gw[e * H_ + h + 3];
    }
  }
  // top-2 (strict > keeps lowest index on ties, matching lax.top_k)
  int i1 = 0;
#pragma unroll
  for (int e = 1; e < E_; e++) if (acc[e] > acc[i1]) i1 = e;
  int i2 = (i1 == 0) ? 1 : 0;
#pragma unroll
  for (int e = 0; e < E_; e++) if (e != i1 && acc[e] > acc[i2]) i2 = e;

  // renormalized top-2 softmax weights: p1/(p1+p2) = 1/(1+exp(l2-l1))
  float p2 = __expf(acc[i2] - acc[i1]);
  float inv = 1.0f / (1.0f + p2);
#pragma unroll
  for (int e = 0; e < E_; e++) {
    float v = (e == i1) ? inv : ((e == i2) ? p2 * inv : 0.0f);
    ew[(size_t)t * E_ + e] = v;
  }
}

// ---------------- Fused SwiGLU MoE expert kernel (M=32 per block) -----------
#define TM 32            // tokens per block (2 x 16-row A tiles share B frags)
#define XS_STRIDE 1032   // 1024 + 8 pad (conflict-free A-fragment ds_load_b128)
#define ACT_STRIDE 2056  // 2048 + 8 pad
#define LDS_BYTES ((TM * XS_STRIDE + TM * ACT_STRIDE) * 2 + TM * E_ * 4 + E_ * 4)

__global__ __launch_bounds__(256, 1) void moe_kernel(
    const unsigned short* __restrict__ xb,
    const unsigned short* __restrict__ w1b,
    const unsigned short* __restrict__ w3b,
    const unsigned short* __restrict__ w2b,
    const float* __restrict__ ew, float* __restrict__ out) {
  extern __shared__ unsigned short smem[];
  unsigned short* xs   = smem;                                   // [TM][XS_STRIDE] bf16 (LDS offset 0)
  unsigned short* acts = smem + TM * XS_STRIDE;                  // [TM][ACT_STRIDE] bf16
  float* ewt = (float*)(smem + TM * XS_STRIDE + TM * ACT_STRIDE);// [TM][E_]
  int* eflag = (int*)(ewt + TM * E_);                            // [E_] active flags

  const int tid  = threadIdx.x;
  const int wave = tid >> 5;
  const int lane = tid & 31;
  const int m    = lane & 15;     // A-fragment row / B-fragment column
  const int khi  = lane >> 4;     // half-wave id
  const int kfo  = khi * 8;       // A K-offset
  const int kb0  = khi * 8;       // B K-offset
  const int tok0 = blockIdx.x * TM;

  // Stage the TM x H bf16 x-tile into LDS with async copies (16B per lane op).
  // xs sits at LDS byte offset 0; all addresses are 16B aligned.
  for (int idx = tid; idx < TM * (H_ / 8); idx += 256) {
    int row = idx >> 7;          // H_/8 == 128 16B-chunks per row
    int c8  = (idx & 127) * 8;
    unsigned lds_off = (unsigned)(row * XS_STRIDE + c8) * 2u;
    unsigned long long ga =
        (unsigned long long)(xb + (size_t)(tok0 + row) * H_ + c8);
    asm volatile("global_load_async_to_lds_b128 %0, %1, off"
                 :: "v"(lds_off), "v"(ga) : "memory");
  }
  if (tid < TM * E_) ewt[tid] = ew[(size_t)tok0 * E_ + tid];
  asm volatile("s_wait_asynccnt 0x0" ::: "memory");
  __syncthreads();

  // Precompute per-expert active flags once (8 threads scan, all read 1 dword).
  if (tid < E_) {
    int f = 0;
#pragma unroll
    for (int r = 0; r < TM; r++) f |= (ewt[r * E_ + tid] != 0.0f) ? 1 : 0;
    eflag[tid] = f;
  }
  __syncthreads();

  // Per-wave output accumulators: 2 tiles x 8 H-subtiles x 8 rows (fp32).
  float oacc0[8][8], oacc1[8][8];
#pragma unroll
  for (int i = 0; i < 8; i++)
#pragma unroll
    for (int j = 0; j < 8; j++) { oacc0[i][j] = 0.0f; oacc1[i][j] = 0.0f; }

  for (int e = 0; e < E_; e++) {
    if (eflag[e] == 0) continue;   // uniform across the workgroup

    // ---- Stage 1: act[TM, I] = silu(x w1^T) * (x w3^T); wave owns 256 I-cols.
    for (int nt = 0; nt < 16; nt++) {
      int icol = wave * 256 + nt * 16 + m;
      const unsigned short* w1p = w1b + ((size_t)e * I_ + icol) * H_ + kb0;
      const unsigned short* w3p = w3b + ((size_t)e * I_ + icol) * H_ + kb0;
      v8f g0 = {}, u0 = {}, g1 = {}, u1 = {};
#pragma unroll 2
      for (int kc = 0; kc < H_; kc += 32) {
        v16bf a0 = load_afrag_lds(xs, XS_STRIDE, m,      kc + kfo);
        v16bf a1 = load_afrag_lds(xs, XS_STRIDE, m + 16, kc + kfo);
        v16bf b1 = load_bfrag_bf16(w1p + kc);
        v16bf b3 = load_bfrag_bf16(w3p + kc);
        g0 = wmma_bf16(a0, b1, g0);
        g1 = wmma_bf16(a1, b1, g1);    // B reused for second A tile
        u0 = wmma_bf16(a0, b3, u0);
        u1 = wmma_bf16(a1, b3, u1);
      }
#pragma unroll
      for (int r = 0; r < 8; r++) {
        acts[(r + khi * 8) * ACT_STRIDE + icol]      = f2bf(silu_f(g0[r]) * u0[r]);
        acts[(16 + r + khi * 8) * ACT_STRIDE + icol] = f2bf(silu_f(g1[r]) * u1[r]);
      }
    }
    __syncthreads();

    // ---- Stage 2: y[TM, H] = act w2^T; wave owns 128 H-cols; fold ew in.
    float ewr0[8], ewr1[8];
#pragma unroll
    for (int r = 0; r < 8; r++) {
      ewr0[r] = ewt[(r + khi * 8) * E_ + e];
      ewr1[r] = ewt[(16 + r + khi * 8) * E_ + e];
    }

    for (int nt = 0; nt < 8; nt++) {
      int hcol = wave * 128 + nt * 16 + m;
      const unsigned short* w2p = w2b + ((size_t)e * H_ + hcol) * I_ + kb0;
      v8f y0 = {}, y1 = {};
#pragma unroll 2
      for (int kc = 0; kc < I_; kc += 32) {
        v16bf a0 = load_afrag_lds(acts, ACT_STRIDE, m,      kc + kfo);
        v16bf a1 = load_afrag_lds(acts, ACT_STRIDE, m + 16, kc + kfo);
        v16bf bf = load_bfrag_bf16(w2p + kc);
        y0 = wmma_bf16(a0, bf, y0);
        y1 = wmma_bf16(a1, bf, y1);
      }
#pragma unroll
      for (int r = 0; r < 8; r++) {
        oacc0[nt][r] += ewr0[r] * y0[r];
        oacc1[nt][r] += ewr1[r] * y1[r];
      }
    }
    __syncthreads();  // acts reused by next expert's stage 1
  }

  // ---- Store: each (token,h) owned by exactly one wave/lane/vgpr.
#pragma unroll
  for (int nt = 0; nt < 8; nt++) {
    int hcol = wave * 128 + nt * 16 + m;
#pragma unroll
    for (int r = 0; r < 8; r++) {
      out[(size_t)(tok0 + r + khi * 8) * H_ + hcol]      = oacc0[nt][r];
      out[(size_t)(tok0 + 16 + r + khi * 8) * H_ + hcol] = oacc1[nt][r];
    }
  }
}

extern "C" void kernel_launch(void* const* d_in, const int* in_sizes, int n_in,
                              void* d_out, int out_size, void* d_ws, size_t ws_size,
                              hipStream_t stream) {
  (void)in_sizes; (void)n_in; (void)out_size; (void)ws_size;
  const float* x      = (const float*)d_in[0];
  const float* gate_w = (const float*)d_in[1];
  const float* w1     = (const float*)d_in[2];
  const float* w3     = (const float*)d_in[3];
  const float* w2     = (const float*)d_in[4];
  float* out = (float*)d_out;

  // Workspace: ew[T*E] fp32 (64 KB), xb bf16 (4 MB), w1b/w3b/w2b bf16 (32 MB ea).
  float* ew = (float*)d_ws;
  unsigned short* xb  = (unsigned short*)((char*)d_ws + (size_t)T_ * E_ * 4);
  unsigned short* w1b = xb + (size_t)T_ * H_;
  unsigned short* w3b = w1b + (size_t)E_ * I_ * H_;
  unsigned short* w2b = w3b + (size_t)E_ * I_ * H_;

  const int wn_blocks = (int)(((size_t)E_ * I_ * H_) / 8 / 256);  // 8192
  const int xn_blocks = (int)(((size_t)T_ * H_) / 8 / 256);       // 1024
  cvt_kernel<<<wn_blocks, 256, 0, stream>>>(w1, w1b);
  cvt_kernel<<<wn_blocks, 256, 0, stream>>>(w3, w3b);
  cvt_kernel<<<wn_blocks, 256, 0, stream>>>(w2, w2b);
  cvt_kernel<<<xn_blocks, 256, 0, stream>>>(x, xb);

  router_kernel<<<T_ / 256, 256, 0, stream>>>(x, gate_w, ew);
  moe_kernel<<<T_ / TM, 256, LDS_BYTES, stream>>>(xb, w1b, w3b, w2b, ew, out);
}
